// MultiHeadAttention_50431505989845
// MI455X (gfx1250) — matrix operand order repaired and checked by reading, not compiled, not run on hardware
//
#include <hip/hip_runtime.h>
#include <cstdint>

// ---------------------------------------------------------------------------
// MultiHeadAttention (softmax over HEADS axis) for MI455X / gfx1250, wave32.
//   1) one-shot f32->bf16 conversion of activations + weights (kills the
//      v_cvt storm inside the GEMM loops, halves GEMM load bytes)
//   2) 4x all-bf16 WMMA GEMMs (q/k/v/out projections)
//   3) fused attention: 8 waves = 8 heads share LDS; K-tile staged in LDS via
//      gfx1250 async global->LDS copies; head-axis softmax through LDS;
//      no global score tensor.
// ---------------------------------------------------------------------------

typedef __bf16 v8bf  __attribute__((ext_vector_type(8)));
typedef __bf16 v16bf __attribute__((ext_vector_type(16)));
typedef float  v8f   __attribute__((ext_vector_type(8)));
typedef int    v4i   __attribute__((ext_vector_type(4)));

static constexpr int DMODEL = 512;
static constexpr int SEQ    = 2048;
static constexpr int BATCH  = 2;
static constexpr int ROWS   = BATCH * SEQ;   // 4096
static constexpr int HEADS  = 8;
static constexpr int DK     = 64;

// --- gfx1250 async global->LDS path (guarded; sync fallback otherwise) -----
#if defined(__has_builtin)
#if __has_builtin(__builtin_amdgcn_global_load_async_to_lds_b128) && \
    __has_builtin(__builtin_amdgcn_s_wait_asynccnt)
#define HAVE_ASYNC_LDS 1
#else
#define HAVE_ASYNC_LDS 0
#endif
#else
#define HAVE_ASYNC_LDS 0
#endif

typedef __attribute__((address_space(1))) v4i as1_v4i;   // global 16B chunk
typedef __attribute__((address_space(3))) v4i as3_v4i;   // LDS 16B chunk

static __device__ __forceinline__ v16bf cat8(v8bf lo, v8bf hi) {
  return __builtin_shufflevector(lo, hi, 0,1,2,3,4,5,6,7,8,9,10,11,12,13,14,15);
}

static __device__ __forceinline__ v8f wmma_bf16(v16bf a, v16bf b, v8f c) {
  // v_wmma_f32_16x16x32_bf16: D = A(16x32) * B(32x16) + C(16x16 f32)
  return __builtin_amdgcn_wmma_f32_16x16x32_bf16(false, a, false, b,
                                                 (short)0, c, false, false);
}

// A-operand gather (16x32 tile): element j -> k = j + 16*lhalf
static __device__ __forceinline__ v16bf load_a_bf16(const __bf16* __restrict__ row,
                                                    int lhalf) {
  const __bf16* p = row + 16 * lhalf;
  v8bf lo = *(const v8bf*)(p);
  v8bf hi = *(const v8bf*)(p + 8);
  return cat8(lo, hi);
}

// B-operand gather (32x16 tile): 16 contiguous elements at (row + 16*lhalf)
static __device__ __forceinline__ v16bf load_b_bf16(const __bf16* __restrict__ p) {
  v8bf lo = *(const v8bf*)(p);
  v8bf hi = *(const v8bf*)(p + 8);
  return cat8(lo, hi);
}

// ---------------------------------------------------------------------------
// One-shot f32 -> bf16 conversion (n must be a multiple of 2048)
// ---------------------------------------------------------------------------
__global__ __launch_bounds__(256) void mha_cvt_bf16_kernel(
    const float* __restrict__ src, __bf16* __restrict__ dst) {
  const int i = (blockIdx.x * 256 + threadIdx.x) * 8;
  const float4 a = *(const float4*)(src + i);
  const float4 b = *(const float4*)(src + i + 4);
  v8f t;
  t[0] = a.x; t[1] = a.y; t[2] = a.z; t[3] = a.w;
  t[4] = b.x; t[5] = b.y; t[6] = b.z; t[7] = b.w;
  *(v8bf*)(dst + i) = __builtin_convertvector(t, v8bf);
}

// ---------------------------------------------------------------------------
// All-bf16 projection GEMM: Y[m,n] = sum_k X[m,k] * W[n,k] + bias[n]
// X:[4096,512] bf16, W:[512,512] bf16 (both contiguous in k -> "NT" GEMM).
// Each wave computes a 16x64 tile (64 WMMAs). Exactly one of outRM (bf16
// row-major), outTR (bf16 transposed [B][D][S]) or outF32 is non-null.
// ---------------------------------------------------------------------------
__global__ __launch_bounds__(128) void mha_proj_kernel(
    const __bf16* __restrict__ X, const __bf16* __restrict__ W,
    const float* __restrict__ bias, __bf16* __restrict__ outRM,
    __bf16* __restrict__ outTR, float* __restrict__ outF32) {
  const int wave  = threadIdx.x >> 5;
  const int lane  = threadIdx.x & 31;
  const int lr    = lane & 15;
  const int lhalf = lane >> 4;
  const int m0 = blockIdx.x * 64 + wave * 16;
  const int n0 = blockIdx.y * 64;

  v8f acc0 = {}, acc1 = {}, acc2 = {}, acc3 = {};
  const __bf16* arow = X + (size_t)(m0 + lr) * DMODEL;
  const __bf16* b0r = W + (size_t)(n0 +  0 + lr) * DMODEL + 16 * lhalf;
  const __bf16* b1r = W + (size_t)(n0 + 16 + lr) * DMODEL + 16 * lhalf;
  const __bf16* b2r = W + (size_t)(n0 + 32 + lr) * DMODEL + 16 * lhalf;
  const __bf16* b3r = W + (size_t)(n0 + 48 + lr) * DMODEL + 16 * lhalf;

#pragma unroll 4
  for (int k0 = 0; k0 < DMODEL; k0 += 32) {
    __builtin_prefetch(arow + k0 + 32, 0, 0);   // global_prefetch_b8
    v16bf a  = load_a_bf16(arow + k0, lhalf);
    v16bf b0 = load_b_bf16(b0r + k0);
    v16bf b1 = load_b_bf16(b1r + k0);
    v16bf b2 = load_b_bf16(b2r + k0);
    v16bf b3 = load_b_bf16(b3r + k0);
    acc0 = wmma_bf16(a, b0, acc0);
    acc1 = wmma_bf16(a, b1, acc1);
    acc2 = wmma_bf16(a, b2, acc2);
    acc3 = wmma_bf16(a, b3, acc3);
  }

#pragma unroll
  for (int t = 0; t < 4; ++t) {
    v8f acc = (t == 0) ? acc0 : (t == 1) ? acc1 : (t == 2) ? acc2 : acc3;
    const int n = n0 + t * 16 + lr;
    const float bb = bias[n];
#pragma unroll
    for (int i = 0; i < 8; ++i) {
      const int m = m0 + i + 8 * lhalf;            // C layout: (i + 8*lhalf, lr)
      const float val = acc[i] + bb;
      if (outF32) {
        outF32[(size_t)m * DMODEL + n] = val;
      } else if (outRM) {
        outRM[(size_t)m * DMODEL + n] = (__bf16)val;
      } else {
        const int bi = m >> 11;                    // m / SEQ
        const int si = m & (SEQ - 1);
        outTR[((size_t)bi * DMODEL + n) * SEQ + si] = (__bf16)val;
      }
    }
  }
}

// ---------------------------------------------------------------------------
// Fused attention with head-axis softmax.
// Block = 256 threads = 8 waves; wave h owns head h. Grid = (S/32, B).
// Per 32-key tile: K-tile (32x512 bf16, 32 KB) async-staged into padded LDS,
// each wave computes its head's 32x32 score tile (8 WMMAs, B-tiles from LDS,
// shared across the two 16-query subtiles), 8-head softmax through LDS, then
// 8 WMMAs of p @ v (v B-tiles reused across subtiles).
// ---------------------------------------------------------------------------
__global__ __launch_bounds__(256) void mha_attn_kernel(
    const __bf16* __restrict__ qb, const __bf16* __restrict__ kb,
    const __bf16* __restrict__ vtb, __bf16* __restrict__ attnb) {
  constexpr int KROW = DMODEL + 8;             // 520 bf16 = 1040 B padded row
  __shared__ __bf16 lds_k[32 * KROW];          // 33,280 B
  __shared__ __bf16 lds_e[1024 * HEADS];       // 16,384 B scaled scores
  __shared__ float  lds_m[1024];               // max over heads
  __shared__ float  lds_r[1024];               // 1 / sum_h exp(s_h - max)

  const int tid   = threadIdx.x;
  const int h     = tid >> 5;                  // head = wave id
  const int lane  = tid & 31;
  const int lr    = lane & 15;
  const int lhalf = lane >> 4;
  const int bi = blockIdx.y;
  const int q0 = blockIdx.x * 32;
  const float scale = 0.125f;                  // 1/sqrt(DK)

  // Persistent q A-tiles: 2 query subtiles x 2 dk chunks
  const __bf16* q0row = qb + (size_t)(bi * SEQ + q0 + lr) * DMODEL + h * DK;
  const __bf16* q1row = q0row + (size_t)16 * DMODEL;
  const v16bf aq00 = load_a_bf16(q0row,      lhalf);
  const v16bf aq01 = load_a_bf16(q0row + 32, lhalf);
  const v16bf aq10 = load_a_bf16(q1row,      lhalf);
  const v16bf aq11 = load_a_bf16(q1row + 32, lhalf);

  v8f acc00 = {}, acc01 = {}, acc02 = {}, acc03 = {};   // subtile 0: 16q x 64dk
  v8f acc10 = {}, acc11 = {}, acc12 = {}, acc13 = {};   // subtile 1

  for (int kt = 0; kt < SEQ / 32; ++kt) {
    // ---- stage K tile (32 rows x 512 cols, contiguous 32 KB) into LDS ----
    {
      const __bf16* gk = kb + (size_t)(bi * SEQ + kt * 32) * DMODEL;
#pragma unroll
      for (int it = 0; it < 8; ++it) {
        const int task = tid + it * 256;       // 0..2047 16-byte chunks
        const int row  = task >> 6;
        const int ch   = task & 63;
        const __bf16* g = gk + (size_t)row * DMODEL + ch * 8;
        __bf16* l = &lds_k[row * KROW + ch * 8];
#if HAVE_ASYNC_LDS
        __builtin_amdgcn_global_load_async_to_lds_b128(
            (as1_v4i*)(uintptr_t)g, (as3_v4i*)(unsigned)(uintptr_t)l, 0, 0);
#else
        *(v8bf*)l = *(const v8bf*)g;
#endif
      }
#if HAVE_ASYNC_LDS
      __builtin_amdgcn_s_wait_asynccnt(0);
#endif
    }
    __syncthreads();

    // ---- scores: 32 queries x 32 keys per head, f32 accumulate -----------
    v8f s00 = {}, s01 = {}, s10 = {}, s11 = {};
    {
      const __bf16* kl0 = &lds_k[( 0 + lr) * KROW + h * DK + 16 * lhalf];
      const __bf16* kl1 = &lds_k[(16 + lr) * KROW + h * DK + 16 * lhalf];
      const v16bf kb00 = load_b_bf16(kl0);        // key half 0, dk chunk 0
      const v16bf kb01 = load_b_bf16(kl0 + 32);   // key half 0, dk chunk 1
      const v16bf kb10 = load_b_bf16(kl1);        // key half 1, dk chunk 0
      const v16bf kb11 = load_b_bf16(kl1 + 32);   // key half 1, dk chunk 1
      s00 = wmma_bf16(aq00, kb00, s00); s00 = wmma_bf16(aq01, kb01, s00);
      s10 = wmma_bf16(aq10, kb00, s10); s10 = wmma_bf16(aq11, kb01, s10);
      s01 = wmma_bf16(aq00, kb10, s01); s01 = wmma_bf16(aq01, kb11, s01);
      s11 = wmma_bf16(aq10, kb10, s11); s11 = wmma_bf16(aq11, kb11, s11);
    }

    // ---- publish scaled scores (bf16) to LDS: elem = q*32 + key ----------
#pragma unroll
    for (int i = 0; i < 8; ++i) {
      const int m = i + 8 * lhalf;
      lds_e[(( 0 + m) * 32 +      lr) * HEADS + h] = (__bf16)(s00[i] * scale);
      lds_e[(( 0 + m) * 32 + 16 + lr) * HEADS + h] = (__bf16)(s01[i] * scale);
      lds_e[((16 + m) * 32 +      lr) * HEADS + h] = (__bf16)(s10[i] * scale);
      lds_e[((16 + m) * 32 + 16 + lr) * HEADS + h] = (__bf16)(s11[i] * scale);
    }
    __syncthreads();

    // ---- softmax over the 8 heads per (q,key) element --------------------
#pragma unroll
    for (int r = 0; r < 4; ++r) {
      const int idx = tid * 4 + r;             // 0..1023
      const v8bf ev = *(const v8bf*)&lds_e[idx * HEADS];
      const v8f  e  = __builtin_convertvector(ev, v8f);
      float mx = e[0];
#pragma unroll
      for (int hh = 1; hh < HEADS; ++hh) mx = fmaxf(mx, e[hh]);
      float d = 0.0f;
#pragma unroll
      for (int hh = 0; hh < HEADS; ++hh) d += __expf(e[hh] - mx);
      lds_m[idx] = mx;
      lds_r[idx] = 1.0f / d;
    }
    __syncthreads();

    // ---- gather this head's probs in WMMA A-layout (2 subtiles) ----------
    v16bf pa0, pa1;
#pragma unroll
    for (int j = 0; j < 16; ++j) {
      const int k  = j + 16 * lhalf;                        // key 0..31
      const int i0 = (lr)      * 32 + k;                    // subtile 0, m = lr
      const int i1 = (16 + lr) * 32 + k;                    // subtile 1
      pa0[j] = (__bf16)(__expf((float)lds_e[i0 * HEADS + h] - lds_m[i0]) * lds_r[i0]);
      pa1[j] = (__bf16)(__expf((float)lds_e[i1 * HEADS + h] - lds_m[i1]) * lds_r[i1]);
    }

    // ---- accumulate p @ v : v B-tiles from global, reused by both subtiles
    {
      const __bf16* vbase = vtb + (size_t)bi * DMODEL * SEQ + kt * 32 + 16 * lhalf;
      const v16bf vb0 = load_b_bf16(vbase + (size_t)(h * DK +  0 + lr) * SEQ);
      const v16bf vb1 = load_b_bf16(vbase + (size_t)(h * DK + 16 + lr) * SEQ);
      const v16bf vb2 = load_b_bf16(vbase + (size_t)(h * DK + 32 + lr) * SEQ);
      const v16bf vb3 = load_b_bf16(vbase + (size_t)(h * DK + 48 + lr) * SEQ);
      acc00 = wmma_bf16(pa0, vb0, acc00);  acc10 = wmma_bf16(pa1, vb0, acc10);
      acc01 = wmma_bf16(pa0, vb1, acc01);  acc11 = wmma_bf16(pa1, vb1, acc11);
      acc02 = wmma_bf16(pa0, vb2, acc02);  acc12 = wmma_bf16(pa1, vb2, acc12);
      acc03 = wmma_bf16(pa0, vb3, acc03);  acc13 = wmma_bf16(pa1, vb3, acc13);
    }
    __syncthreads();   // protect lds_e / lds_k against next iteration
  }

  // ---- store attn tiles (bf16, row-major [4096, 512]) ---------------------
#pragma unroll
  for (int t = 0; t < 4; ++t) {
    v8f a0 = (t == 0) ? acc00 : (t == 1) ? acc01 : (t == 2) ? acc02 : acc03;
    v8f a1 = (t == 0) ? acc10 : (t == 1) ? acc11 : (t == 2) ? acc12 : acc13;
    const int col = h * DK + t * 16 + lr;
#pragma unroll
    for (int i = 0; i < 8; ++i) {
      const size_t row0 = (size_t)(bi * SEQ + q0 + i + 8 * lhalf);
      attnb[row0 * DMODEL + col]        = (__bf16)a0[i];
      attnb[(row0 + 16) * DMODEL + col] = (__bf16)a1[i];
    }
  }
}

// ---------------------------------------------------------------------------
extern "C" void kernel_launch(void* const* d_in, const int* in_sizes, int n_in,
                              void* d_out, int out_size, void* d_ws, size_t ws_size,
                              hipStream_t stream) {
  (void)in_sizes; (void)n_in; (void)out_size; (void)ws_size;
  const float* Q   = (const float*)d_in[0];
  const float* K   = (const float*)d_in[1];
  const float* V   = (const float*)d_in[2];
  const float* w_q = (const float*)d_in[3];
  const float* b_q = (const float*)d_in[4];
  const float* w_k = (const float*)d_in[5];
  const float* b_k = (const float*)d_in[6];
  const float* w_v = (const float*)d_in[7];
  const float* b_v = (const float*)d_in[8];
  const float* w_o = (const float*)d_in[9];
  const float* b_o = (const float*)d_in[10];
  float* out = (float*)d_out;

  // Workspace carve-out (bf16 elements):
  const size_t ACT = (size_t)ROWS * DMODEL;    // 2M elems
  const size_t WEL = (size_t)DMODEL * DMODEL;  // 256K elems
  __bf16* Xq  = (__bf16*)d_ws;                 // bf16 copies of inputs
  __bf16* Xk  = Xq  + ACT;
  __bf16* Xv  = Xk  + ACT;
  __bf16* Wq  = Xv  + ACT;                     // bf16 copies of weights
  __bf16* Wk  = Wq  + WEL;
  __bf16* Wv  = Wk  + WEL;
  __bf16* Wo  = Wv  + WEL;
  __bf16* qb    = Wo  + WEL;                   // projected q (row-major)
  __bf16* kbuf  = qb  + ACT;                   // projected k (row-major)
  __bf16* vtb   = kbuf + ACT;                  // projected v (transposed [B][D][S])
  __bf16* attnb = vtb + (size_t)BATCH * DMODEL * SEQ;

  // 1) f32 -> bf16 conversions
  const int gAct = (int)(ACT / 2048);          // 1024 blocks
  const int gW   = (int)(WEL / 2048);          // 128 blocks
  mha_cvt_bf16_kernel<<<gAct, 256, 0, stream>>>(Q,   Xq);
  mha_cvt_bf16_kernel<<<gAct, 256, 0, stream>>>(K,   Xk);
  mha_cvt_bf16_kernel<<<gAct, 256, 0, stream>>>(V,   Xv);
  mha_cvt_bf16_kernel<<<gW,   256, 0, stream>>>(w_q, Wq);
  mha_cvt_bf16_kernel<<<gW,   256, 0, stream>>>(w_k, Wk);
  mha_cvt_bf16_kernel<<<gW,   256, 0, stream>>>(w_v, Wv);
  mha_cvt_bf16_kernel<<<gW,   256, 0, stream>>>(w_o, Wo);

  // 2) projections
  dim3 pgrid(ROWS / 64, DMODEL / 64);
  mha_proj_kernel<<<pgrid, 128, 0, stream>>>(Xq, Wq, b_q, qb,   nullptr, nullptr);
  mha_proj_kernel<<<pgrid, 128, 0, stream>>>(Xk, Wk, b_k, kbuf, nullptr, nullptr);
  mha_proj_kernel<<<pgrid, 128, 0, stream>>>(Xv, Wv, b_v, nullptr, vtb,  nullptr);

  // 3) fused attention (softmax over heads)
  mha_attn_kernel<<<dim3(SEQ / 32, BATCH), 256, 0, stream>>>(qb, kbuf, vtb, attnb);

  // 4) output projection -> f32
  mha_proj_kernel<<<pgrid, 128, 0, stream>>>(attnb, Wo, b_o, nullptr, nullptr, out);
}
